// Qwen3Attention_19000935318014
// MI455X (gfx1250) — compile-verified, matrix operand order checked by reading, not run on hardware
//
#include <hip/hip_runtime.h>
#include <hip/hip_bf16.h>

// ---------------------------------------------------------------------------
// Qwen3 attention block for MI455X (gfx1250, wave32, WMMA).
// All matrix contractions use V_WMMA_F32_16X16X32_F16.
// ---------------------------------------------------------------------------

typedef _Float16 f16;
typedef __attribute__((ext_vector_type(16))) _Float16 v16h;
typedef __attribute__((ext_vector_type(8)))  _Float16 v8h;
typedef __attribute__((ext_vector_type(8)))  float    v8f;

#define S_LEN 2048
#define HID   2048
#define NH    16
#define HD    128

// ---- WMMA fragment helpers -------------------------------------------------
// 16-bit A-matrix 16x32 layout (ISA 7.12.2):
//   lanes 0-15 : VGPR0-3 K=0..7,  VGPR4-7 K=16..23   (rows M = lane)
//   lanes16-31 : VGPR0-3 K=8..15, VGPR4-7 K=24..31   (rows M = lane-16)
// => per lane: 16B chunk at K = hi*8 and 16B chunk at K = hi*8+16.
// B fragments ([N,K] row-major, i.e. KxN column-major) use the identical
// pattern with N in place of M.
static __device__ __forceinline__ v16h frag_ld(const f16* p, int hi) {
    const v8h c0 = *reinterpret_cast<const v8h*>(p + hi * 8);
    const v8h c1 = *reinterpret_cast<const v8h*>(p + hi * 8 + 16);
    return __builtin_shufflevector(c0, c1, 0,1,2,3,4,5,6,7,8,9,10,11,12,13,14,15);
}

static __device__ __forceinline__ v8f wmma_f16(v16h a, v16h b, v8f c) {
    return __builtin_amdgcn_wmma_f32_16x16x32_f16(false, a, false, b,
                                                  (short)0, c, false, false);
}

// ---- 1) global absmax reduce ----------------------------------------------
__global__ void absmax_kernel(const float* __restrict__ x, int n,
                              float* __restrict__ out) {
    int lane = threadIdx.x & 31;
    float m = 0.f;
    for (int i = blockIdx.x * blockDim.x + threadIdx.x; i < n;
         i += gridDim.x * blockDim.x)
        m = fmaxf(m, fabsf(x[i]));
#pragma unroll
    for (int off = 16; off >= 1; off >>= 1)
        m = fmaxf(m, __shfl_xor(m, off));
    if (lane == 0)  // |x| >= 0: uint compare == float compare
        atomicMax(reinterpret_cast<unsigned int*>(out), __float_as_uint(m));
}

// ---- 2) 16-bit fake-quant of hidden -> f16 A matrix ------------------------
__global__ void qdq16_kernel(const float* __restrict__ x, int n,
                             const float* __restrict__ amax,
                             f16* __restrict__ y) {
    int i = blockIdx.x * blockDim.x + threadIdx.x;
    if (i >= n) return;
    float s = fmaxf(amax[0] / 32767.f, 1e-12f);
    float q = fminf(fmaxf(rintf(x[i] / s), -32768.f), 32767.f) * s;
    y[i] = (f16)q;
}

// ---- 3) block-32 weight fake-quant (qmax = 7) -> f16 [O,K] -----------------
__global__ void wprep_kernel(const float* __restrict__ w, int nblk,
                             f16* __restrict__ out) {
    int t = blockIdx.x * blockDim.x + threadIdx.x;
    if (t >= nblk) return;
    const float* src = w + (size_t)t * 32;
    f16* dst = out + (size_t)t * 32;
    float amax = 0.f;
#pragma unroll
    for (int i = 0; i < 32; ++i) amax = fmaxf(amax, fabsf(src[i]));
    float s = fmaxf(amax / 7.f, 1e-12f);
#pragma unroll
    for (int i = 0; i < 32; ++i) {
        float q = fminf(fmaxf(rintf(src[i] / s), -8.f), 7.f) * s;
        dst[i] = (f16)q;
    }
}

// ---- 4) WMMA GEMM: C[M,N] = A[M,K] * B[N,K]^T ------------------------------
// wave tile 16(M) x 64(N); block = 8 waves stacked on M -> 128x64.
__global__ void gemm_wmma_kernel(const f16* __restrict__ A,
                                 const f16* __restrict__ B,
                                 float* __restrict__ C,
                                 int M, int N, int K) {
    const int wave = threadIdx.x >> 5;
    const int lane = threadIdx.x & 31;
    const int hi   = lane >> 4;
    const int l16  = lane & 15;
    const int m0 = blockIdx.y * 128 + wave * 16;
    const int n0 = blockIdx.x * 64;

    v8f acc[4] = {};
    const f16* arow = A + (size_t)(m0 + l16) * K;
    for (int k0 = 0; k0 < K; k0 += 32) {
        v16h a = frag_ld(arow + k0, hi);
#pragma unroll
        for (int i = 0; i < 4; ++i) {
            v16h b = frag_ld(B + (size_t)(n0 + i * 16 + l16) * K + k0, hi);
            acc[i] = wmma_f16(a, b, acc[i]);
        }
    }
#pragma unroll
    for (int i = 0; i < 4; ++i)
#pragma unroll
        for (int r = 0; r < 8; ++r) {
            int row = m0 + r + 8 * hi;         // C layout: VGPR r, half hi
            int col = n0 + i * 16 + l16;
            C[(size_t)row * N + col] = acc[i][r];
        }
}

// ---- 5) per-(s,h) head post-processing -------------------------------------
// one wave per (s,h): RMSNorm, RoPE, 8-bit per-row fake-quant of K and V.
// outputs: Qh,Kh f16 [H][S][HD];  Vt f16 [H][HD][S] (pre-transposed for PV).
__global__ void posthead_kernel(const float* __restrict__ Qr,
                                const float* __restrict__ Kr,
                                const float* __restrict__ Vr,
                                const float* __restrict__ sinp,
                                const float* __restrict__ cosp,
                                const float* __restrict__ qnw,
                                const float* __restrict__ knw,
                                f16* __restrict__ Qh,
                                f16* __restrict__ Kh,
                                f16* __restrict__ Vt) {
    const int wave = threadIdx.x >> 5;
    const int lane = threadIdx.x & 31;
    const int wid  = blockIdx.x * 8 + wave;
    const int s = wid / NH;
    const int h = wid % NH;
    if (s >= S_LEN) return;

    const size_t base = (size_t)s * (NH * HD) + (size_t)h * HD;
    float q[4], k[4], v[4];
#pragma unroll
    for (int i = 0; i < 4; ++i) {
        int d = lane + 32 * i;
        q[i] = Qr[base + d];
        k[i] = Kr[base + d];
        v[i] = Vr[base + d];
    }
    // RMSNorm over HD=128 (full-wave reduction)
    float sq = 0.f, sk = 0.f;
#pragma unroll
    for (int i = 0; i < 4; ++i) { sq += q[i] * q[i]; sk += k[i] * k[i]; }
#pragma unroll
    for (int off = 16; off >= 1; off >>= 1) {
        sq += __shfl_xor(sq, off);
        sk += __shfl_xor(sk, off);
    }
    float iq = rsqrtf(sq * (1.f / HD) + 1e-6f);
    float ik = rsqrtf(sk * (1.f / HD) + 1e-6f);
#pragma unroll
    for (int i = 0; i < 4; ++i) {
        int d = lane + 32 * i;
        q[i] = q[i] * iq * qnw[d];
        k[i] = k[i] * ik * knw[d];
    }
    // RoPE: elements 0,1 are d<64; 2,3 are d>=64
    float qr4[4] = { -q[2], -q[3], q[0], q[1] };
    float kr4[4] = { -k[2], -k[3], k[0], k[1] };
#pragma unroll
    for (int i = 0; i < 4; ++i) {
        int d = lane + 32 * i;
        float c = cosp[(size_t)s * HD + d];
        float sn = sinp[(size_t)s * HD + d];
        q[i] = q[i] * c + qr4[i] * sn;
        k[i] = k[i] * c + kr4[i] * sn;
    }
    // 8-bit per-row fake-quant of K and V (mirrors reference _qdq(.,8))
    float ak = 0.f, av = 0.f;
#pragma unroll
    for (int i = 0; i < 4; ++i) { ak = fmaxf(ak, fabsf(k[i])); av = fmaxf(av, fabsf(v[i])); }
#pragma unroll
    for (int off = 16; off >= 1; off >>= 1) {
        ak = fmaxf(ak, __shfl_xor(ak, off));
        av = fmaxf(av, __shfl_xor(av, off));
    }
    float s8k = fmaxf(ak / 127.f, 1e-12f);
    float s8v = fmaxf(av / 127.f, 1e-12f);
#pragma unroll
    for (int i = 0; i < 4; ++i) {
        int d = lane + 32 * i;
        float kq = fminf(fmaxf(rintf(k[i] / s8k), -128.f), 127.f) * s8k;
        float vq = fminf(fmaxf(rintf(v[i] / s8v), -128.f), 127.f) * s8v;
        size_t qi = ((size_t)h * S_LEN + s) * HD + d;
        Qh[qi] = (f16)q[i];
        Kh[qi] = (f16)kq;
        Vt[((size_t)h * HD + d) * S_LEN + s] = (f16)vq;  // transposed
    }
}

// ---- 6) flash attention, one wave per (head, 16-query tile) ----------------
__global__ void attn_kernel(const f16* __restrict__ Qh,
                            const f16* __restrict__ Kh,
                            const f16* __restrict__ Vt,
                            f16* __restrict__ Yo) {
    __shared__ __align__(16) f16 lds[8][16 * 32];   // per-wave P staging
    const int wave = threadIdx.x >> 5;
    const int lane = threadIdx.x & 31;
    const int hi   = lane >> 4;
    const int l16  = lane & 15;
    const int wid  = blockIdx.x * 8 + wave;
    const int h  = wid >> 7;         // 2048 waves = 16 heads * 128 q-tiles
    const int qt = wid & 127;
    const int q0 = qt * 16;

    // load the 16x128 Q tile as 4 A-fragments (K chunks of 32)
    v16h qa[4];
    const f16* qrow = Qh + ((size_t)h * S_LEN + q0 + l16) * HD;
#pragma unroll
    for (int c = 0; c < 4; ++c) qa[c] = frag_ld(qrow + c * 32, hi);

    v8f yacc[8] = {};
    float mrow[8], lrow[8];
#pragma unroll
    for (int r = 0; r < 8; ++r) { mrow[r] = -1e30f; lrow[r] = 0.f; }

    const float inv_scaling = 11.31370849898476f;   // / HD^-0.5 (ref divides by denom)
    const int nsteps = (q0 + 16 + 31) >> 5;         // causal: keys <= q0+15

    for (int step = 0; step < nsteps; ++step) {
        const int j0 = step * 32;
        // ---- scores: 16x32 via 2 C tiles, 4 K-chunks each (8 WMMAs)
        v8f sc[2] = {};
#pragma unroll
        for (int t = 0; t < 2; ++t) {
            const f16* krow = Kh + ((size_t)h * S_LEN + j0 + t * 16 + l16) * HD;
#pragma unroll
            for (int c = 0; c < 4; ++c)
                sc[t] = wmma_f16(qa[c], frag_ld(krow + c * 32, hi), sc[t]);
        }
        // ---- scale + causal mask + online softmax per row group r
#pragma unroll
        for (int r = 0; r < 8; ++r) {
            const int row = q0 + r + 8 * hi;
            float s0 = sc[0][r] * inv_scaling;
            float s1 = sc[1][r] * inv_scaling;
            if (j0 + l16      > row) s0 = -1e30f;
            if (j0 + 16 + l16 > row) s1 = -1e30f;
            float mx = fmaxf(s0, s1);
#pragma unroll
            for (int off = 8; off >= 1; off >>= 1)   // 16-lane half reduce
                mx = fmaxf(mx, __shfl_xor(mx, off));
            float mnew = fmaxf(mrow[r], mx);
            float corr = __expf(mrow[r] - mnew);
            float p0 = __expf(s0 - mnew);
            float p1 = __expf(s1 - mnew);
            float ps = p0 + p1;
#pragma unroll
            for (int off = 8; off >= 1; off >>= 1)
                ps += __shfl_xor(ps, off);
            lrow[r] = lrow[r] * corr + ps;
            mrow[r] = mnew;
#pragma unroll
            for (int t = 0; t < 8; ++t) yacc[t][r] *= corr;
            const int prow = r + 8 * hi;             // C-layout row of this VGPR
            lds[wave][prow * 32 + l16]      = (f16)p0;
            lds[wave][prow * 32 + 16 + l16] = (f16)p1;
        }
        // ---- reload P as a 16x32 A-fragment from LDS (ds_load_b128)
        v16h pa = frag_ld(&lds[wave][l16 * 32], hi);
        // ---- PV: y[16x128] += P[16x32] * V[32x128]  (8 WMMAs)
#pragma unroll
        for (int t = 0; t < 8; ++t) {
            const f16* vrow = Vt + ((size_t)h * HD + t * 16 + l16) * S_LEN + j0;
            yacc[t] = wmma_f16(pa, frag_ld(vrow, hi), yacc[t]);
        }
    }
    // ---- epilogue: normalize and write [S, H*HD] f16
#pragma unroll
    for (int t = 0; t < 8; ++t)
#pragma unroll
        for (int r = 0; r < 8; ++r) {
            int row = q0 + r + 8 * hi;
            int col = h * HD + t * 16 + l16;
            Yo[(size_t)row * (NH * HD) + col] = (f16)(yacc[t][r] / lrow[r]);
        }
}

// ---------------------------------------------------------------------------
extern "C" void kernel_launch(void* const* d_in, const int* in_sizes, int n_in,
                              void* d_out, int out_size, void* d_ws, size_t ws_size,
                              hipStream_t stream) {
    const float* hidden = (const float*)d_in[0];
    const float* sinp   = (const float*)d_in[1];
    const float* cosp   = (const float*)d_in[2];
    // d_in[3] causal_mask: causality is computed arithmetically
    const float* wq     = (const float*)d_in[4];
    const float* wk     = (const float*)d_in[5];
    const float* wv     = (const float*)d_in[6];
    const float* wo     = (const float*)d_in[7];
    const float* qnw    = (const float*)d_in[8];
    const float* knw    = (const float*)d_in[9];

    const size_t NEL   = (size_t)S_LEN * HID;        // 4M elements
    const size_t F16SZ = NEL * sizeof(f16);          // 8 MB
    const size_t F32SZ = NEL * sizeof(float);        // 16 MB

    char* ws = (char*)d_ws;
    size_t off = 0;
    float* amax = (float*)(ws + off); off += 256;
    f16* hq  = (f16*)(ws + off); off += F16SZ;
    f16* wqp = (f16*)(ws + off); off += F16SZ;
    f16* wkp = (f16*)(ws + off); off += F16SZ;
    f16* wvp = (f16*)(ws + off); off += F16SZ;
    f16* wop = (f16*)(ws + off); off += F16SZ;
    float* Qr = (float*)(ws + off); off += F32SZ;
    float* Kr = (float*)(ws + off); off += F32SZ;
    float* Vr = (float*)(ws + off); off += F32SZ;
    f16* QhT = (f16*)(ws + off); off += F16SZ;
    f16* KhT = (f16*)(ws + off); off += F16SZ;
    f16* VtT = (f16*)(ws + off); off += F16SZ;
    f16* Yo  = (f16*)(ws + off); off += F16SZ;
    if (off > ws_size) return;  // workspace too small: nothing safe to do

    float* out = (float*)d_out;

    // 1) absmax + 16-bit QDQ of hidden -> f16
    hipMemsetAsync(amax, 0, sizeof(float), stream);
    absmax_kernel<<<512, 256, 0, stream>>>(hidden, (int)NEL, amax);
    qdq16_kernel<<<(int)((NEL + 255) / 256), 256, 0, stream>>>(hidden, (int)NEL, amax, hq);

    // 2) weight block-quant -> f16 [O,K]
    const int nblk = (int)(NEL / 32);
    wprep_kernel<<<(nblk + 255) / 256, 256, 0, stream>>>(wq, nblk, wqp);
    wprep_kernel<<<(nblk + 255) / 256, 256, 0, stream>>>(wk, nblk, wkp);
    wprep_kernel<<<(nblk + 255) / 256, 256, 0, stream>>>(wv, nblk, wvp);
    wprep_kernel<<<(nblk + 255) / 256, 256, 0, stream>>>(wo, nblk, wop);

    // 3) QKV projections (WMMA)
    dim3 ggrid(HID / 64, S_LEN / 128);
    gemm_wmma_kernel<<<ggrid, 256, 0, stream>>>(hq, wqp, Qr, S_LEN, HID, HID);
    gemm_wmma_kernel<<<ggrid, 256, 0, stream>>>(hq, wkp, Kr, S_LEN, HID, HID);
    gemm_wmma_kernel<<<ggrid, 256, 0, stream>>>(hq, wvp, Vr, S_LEN, HID, HID);

    // 4) RMSNorm + RoPE + 8-bit KV quant; V pre-transposed
    posthead_kernel<<<(S_LEN * NH) / 8, 256, 0, stream>>>(
        Qr, Kr, Vr, sinp, cosp, qnw, knw, QhT, KhT, VtT);

    // 5) causal flash attention (WMMA QK^T and PV)
    attn_kernel<<<(NH * (S_LEN / 16)) / 8, 256, 0, stream>>>(QhT, KhT, VtT, Yo);

    // 6) output projection (WMMA) -> f32 d_out
    gemm_wmma_kernel<<<ggrid, 256, 0, stream>>>(Yo, wop, out, S_LEN, HID, HID);
}
